// Sage_3015067042505
// MI455X (gfx1250) — compile-verified, matrix operand order checked by reading, not run on hardware
//
#include <hip/hip_runtime.h>
#include <math.h>

// ---------------- problem constants ----------------
#define N_NODES  16384
#define N_EDGES  49152
#define N_GRAPHS 128
#define NPG      128
#define F_IN     6
#define H        64
#define BN_EPS   1e-3f
#define EPSV     1e-5f
#define LRELU    0.15f
#define PI_F     3.14159265358979323846f

typedef __attribute__((ext_vector_type(16))) __bf16 v16bf;
typedef __attribute__((ext_vector_type(8)))  float  v8f;
typedef __attribute__((ext_vector_type(4)))  unsigned u32x4;
typedef __attribute__((ext_vector_type(8)))  int      i32x8;
typedef __attribute__((ext_vector_type(4)))  int      i32x4;

#if __has_builtin(__builtin_amdgcn_tensor_load_to_lds) && __has_builtin(__builtin_amdgcn_s_wait_tensorcnt)
#define HAVE_TDM 1
#else
#define HAVE_TDM 0
#endif

// ---- bf16 helpers (round to nearest even) ----
__device__ __forceinline__ unsigned bfr(float f) {
  unsigned u = __builtin_bit_cast(unsigned, f);
  return (u + 0x7FFFu + ((u >> 16) & 1u)) >> 16;
}
__device__ __forceinline__ unsigned packbf(float lo, float hi) {
  return bfr(lo) | (bfr(hi) << 16);
}
__device__ __forceinline__ unsigned short bf16u(float f) { return (unsigned short)bfr(f); }
__device__ __forceinline__ __bf16 bflo(unsigned p) {
  return __builtin_bit_cast(__bf16, (unsigned short)(p & 0xffffu));
}
__device__ __forceinline__ __bf16 bfhi(unsigned p) {
  return __builtin_bit_cast(__bf16, (unsigned short)(p >> 16));
}
// K-pair base for WMMA 16x16x32 bf16 A/B fragments (ISA 7.12.2)
__device__ __forceinline__ int kpair(int j, int half) {
  return ((j < 4) ? (2 * j) : (2 * (j - 4) + 16)) + half * 8;
}

// =====================================================================
// Generic GEMM:  C[MxN] = act( A[MxK] @ W[KxN] + bias[N] )
// f32 in/out; bf16 pairs staged in LDS; WMMA f32 accumulate.
// Block = 128 threads (4 waves). Tile: 16(M) x 64(N), K-step 32.
// LDS holds bf16 PAIRS along K (uint): exactly what the frag build reads.
// Fast path (branch-free float4 staging) when no K tail; guarded scalar
// staging only on the last partial K step.
// act: 0=none, 1=relu, 2=leaky(0.15)
// =====================================================================
__global__ void gemm_bias_act(const float* __restrict__ A,
                              const float* __restrict__ W,
                              const float* __restrict__ bias,
                              float* __restrict__ C,
                              int M, int N, int K, int act) {
  __shared__ unsigned PA[16][16];   // [m][kpair]  1 KB
  __shared__ unsigned PB[16][64];   // [kpair][n]  4 KB
  const int mBase = blockIdx.y * 16;
  const int nBase = blockIdx.x * 64;
  const int tid = threadIdx.x;
  const int w = tid >> 5, lane = tid & 31;
  const int half = lane >> 4, nn = lane & 15;

  v8f c = {};
  for (int k0 = 0; k0 < K; k0 += 32) {
    const bool ktail = (k0 + 32 > K);
    __syncthreads();
    if (!ktail && ((K & 3) == 0) && (mBase + 16 <= M)) {
      // ---- fast A staging: one float4 per thread ----
      int m = tid >> 3, q = tid & 7;
      float4 va = *(const float4*)(A + (size_t)(mBase + m) * K + k0 + 4 * q);
      PA[m][2 * q]     = packbf(va.x, va.y);
      PA[m][2 * q + 1] = packbf(va.z, va.w);
    } else {
      for (int idx = tid; idx < 256; idx += 128) {
        int m = idx >> 4, p = idx & 15;
        int row = mBase + m, k = k0 + 2 * p;
        float v0 = (row < M && k < K)     ? A[(size_t)row * K + k]     : 0.f;
        float v1 = (row < M && k + 1 < K) ? A[(size_t)row * K + k + 1] : 0.f;
        PA[m][p] = packbf(v0, v1);
      }
    }
    if (!ktail && ((N & 3) == 0) && (nBase + 64 <= N)) {
      // ---- fast B staging: pair of float4 rows per entry-quad ----
      for (int q = tid; q < 256; q += 128) {
        int pr = q >> 4;            // k-pair row 0..15
        int c4 = (q & 15) * 4;      // n offset 0..60
        const float* b0 = W + (size_t)(k0 + 2 * pr) * N + nBase + c4;
        float4 r0 = *(const float4*)b0;
        float4 r1 = *(const float4*)(b0 + N);
        PB[pr][c4 + 0] = packbf(r0.x, r1.x);
        PB[pr][c4 + 1] = packbf(r0.y, r1.y);
        PB[pr][c4 + 2] = packbf(r0.z, r1.z);
        PB[pr][c4 + 3] = packbf(r0.w, r1.w);
      }
    } else {
      for (int idx = tid; idx < 1024; idx += 128) {
        int pr = idx >> 6, cc = idx & 63;
        int kr = k0 + 2 * pr, col = nBase + cc;
        float v0 = (kr < K && col < N)     ? W[(size_t)kr * N + col]       : 0.f;
        float v1 = (kr + 1 < K && col < N) ? W[(size_t)(kr + 1) * N + col] : 0.f;
        PB[pr][cc] = packbf(v0, v1);
      }
    }
    if (tid == 0 && (k0 + 32) < K)  // gfx1250 global_prefetch of next W tile
      __builtin_prefetch(W + (size_t)(k0 + 32) * N + nBase, 0, 1);
    __syncthreads();

    v16bf a, b;
    const int nc = w * 16 + nn;
#pragma unroll
    for (int j = 0; j < 8; ++j) {
      int kb = kpair(j, half);
      unsigned pa = PA[nn][kb >> 1];
      unsigned pb = PB[kb >> 1][nc];
      a[2 * j]     = bflo(pa);
      a[2 * j + 1] = bfhi(pa);
      b[2 * j]     = bflo(pb);
      b[2 * j + 1] = bfhi(pb);
    }
    c = __builtin_amdgcn_wmma_f32_16x16x32_bf16(false, a, false, b,
                                                (short)0, c, false, false);
  }

  const int n = nBase + w * 16 + nn;
  if (n < N) {
    float bv = bias ? bias[n] : 0.f;
#pragma unroll
    for (int r = 0; r < 8; ++r) {
      int m = mBase + r + half * 8;
      if (m < M) {
        float v = c[r] + bv;
        if (act == 1)      v = fmaxf(v, 0.f);
        else if (act == 2) v = (v > 0.f) ? v : LRELU * v;
        C[(size_t)m * N + n] = v;
      }
    }
  }
}

#if HAVE_TDM
// Issue a TDM 2D-tile load: 64 rows x 64 bf16 cols out of a 64 x 4096 bf16
// tensor (row stride 4096 elems) into LDS at lds_addr. One call per wave.
__device__ __forceinline__ void tdm_load_tile(unsigned lds_addr, const void* gaddr) {
  unsigned long long ga = (unsigned long long)(uintptr_t)gaddr;
  u32x4 g0;
  g0[0] = 1u;                                             // count=1 (valid user D#)
  g0[1] = lds_addr;                                       // lds_addr
  g0[2] = (unsigned)(ga & 0xffffffffu);                   // global_addr[31:0]
  g0[3] = (unsigned)((ga >> 32) & 0x1ffffffu) | (2u << 30); // addr[56:32] | type=2
  i32x8 g1;
  g1[0] = 0x00010000;   // workgroup_mask=0, data_size=1 (2 bytes)
  g1[1] = 0x10000000;   // tensor_dim0 = 4096 (bits 79:48 -> low16 here)
  g1[2] = 0x00400000;   // tensor_dim1 = 64   (bits 111:80 -> low16 here)
  g1[3] = 0x00400000;   // tile_dim0 = 64     (bits 127:112)
  g1[4] = 64;           // tile_dim1 = 64
  g1[5] = 4096;         // tensor_dim0_stride = 4096 (lo32)
  g1[6] = 0;
  g1[7] = 0;
  i32x4 z4 = {0, 0, 0, 0};
#if defined(__clang_major__) && (__clang_major__ >= 23)
  i32x8 z8 = {0, 0, 0, 0, 0, 0, 0, 0};
  __builtin_amdgcn_tensor_load_to_lds(g0, g1, z4, z4, z8, 0);
#else
  __builtin_amdgcn_tensor_load_to_lds(g0, g1, z4, z4, 0);
#endif
}
#endif

// =====================================================================
// Fused edge-kernel GEMM + einsum (Wk4 pre-converted to bf16):
//   kflat[e, f*64+g] = k3[e] @ Wk4[:, f*64+g] + bk4   (never materialized)
//   msg[e,g] = sum_f h[send[e],f] * kflat[e, f*64+g]
//   hdst[recv[e], g] += msg[e,g] * mask2[e]
// One block (4 waves) per 16 edges; 64 f-chunks of 64 columns.
// Chunk staging double-buffered through the Tensor Data Mover.
// =====================================================================
__global__ void kmat_msg_kernel(const float* __restrict__ k3,
                                const float* __restrict__ h,
                                const unsigned short* __restrict__ Wk4bf,
                                const float* __restrict__ bk4,
                                const int* __restrict__ ei,
                                const float* __restrict__ mask2,
                                float* __restrict__ hdst) {
  __shared__ __bf16 As[16][64];      // k3 tile       2 KB
  __shared__ float  Hs[16][64];      // h[send] tile  4 KB
  __shared__ __bf16 Bs[2][64][64];   // Wk4 chunks   16 KB (double buffer)
  const int e0 = blockIdx.x * 16;
  const int tid = threadIdx.x;
  const int w = tid >> 5, lane = tid & 31;
  const int half = lane >> 4, nn = lane & 15;

  for (int idx = tid; idx < 16 * 64; idx += 128) {
    int i = idx >> 6, f = idx & 63;
    int e = e0 + i;
    As[i][f] = __builtin_bit_cast(__bf16, bf16u(k3[(size_t)e * 64 + f]));
    int s = ei[2 * e + 0];
    Hs[i][f] = h[(size_t)s * 64 + f];
  }
#if HAVE_TDM
  if (w == 0)
    tdm_load_tile((unsigned)(uintptr_t)&Bs[0][0][0], Wk4bf);
#endif
  __syncthreads();

  // A fragments (K = 64 -> two k-steps), identical for all 4 waves
  v16bf a0, a1;
#pragma unroll
  for (int j = 0; j < 8; ++j) {
    int kb = kpair(j, half);
    a0[2 * j]     = As[nn][kb];
    a0[2 * j + 1] = As[nn][kb + 1];
    a1[2 * j]     = As[nn][kb + 32];
    a1[2 * j + 1] = As[nn][kb + 33];
  }

  const int g = w * 16 + nn;
  v8f macc = {};
  for (int f = 0; f < 64; ++f) {
    const int buf = f & 1;
#if HAVE_TDM
    if (w == 0) __builtin_amdgcn_s_wait_tensorcnt(0);
    __syncthreads();                    // chunk f resident for all waves
    if (w == 0 && f + 1 < 64)
      tdm_load_tile((unsigned)(uintptr_t)&Bs[buf ^ 1][0][0],
                    Wk4bf + (size_t)(f + 1) * 64);
#else
    __syncthreads();
    {  // plain bf16 copy (uint = 2 bf16), 16 per thread
      const unsigned* src = (const unsigned*)(Wk4bf + (size_t)f * 64);
      unsigned* dst = (unsigned*)&Bs[buf][0][0];
      for (int idx = tid; idx < 2048; idx += 128) {
        int kk = idx >> 5, jj = idx & 31;
        dst[kk * 32 + jj] = src[kk * 2048 + jj];  // row stride 4096 bf16
      }
    }
    __syncthreads();
#endif
    v16bf b0, b1;
#pragma unroll
    for (int j = 0; j < 8; ++j) {
      int kb = kpair(j, half);
      b0[2 * j]     = Bs[buf][kb][g];
      b0[2 * j + 1] = Bs[buf][kb + 1][g];
      b1[2 * j]     = Bs[buf][kb + 32][g];
      b1[2 * j + 1] = Bs[buf][kb + 33][g];
    }
    v8f c = {};
    c = __builtin_amdgcn_wmma_f32_16x16x32_bf16(false, a0, false, b0,
                                                (short)0, c, false, false);
    c = __builtin_amdgcn_wmma_f32_16x16x32_bf16(false, a1, false, b1,
                                                (short)0, c, false, false);
    float bb = bk4[f * 64 + g];
#pragma unroll
    for (int r = 0; r < 8; ++r) {
      int m = r + half * 8;
      macc[r] += Hs[m][f] * (c[r] + bb);
    }
#if !HAVE_TDM
#else
    __syncthreads();   // all waves done with Bs[buf] before TDM refills it
#endif
  }

#pragma unroll
  for (int r = 0; r < 8; ++r) {
    int m = r + half * 8;
    int e = e0 + m;
    float fm = mask2[e];
    if (fm != 0.f) {
      int rv = ei[2 * e + 1];
      atomicAdd(&hdst[(size_t)rv * 64 + g], macc[r] * fm);
    }
  }
}

// ---------------- f32 -> bf16 bulk convert (for Wk4) ----------------------
__global__ void cvt_bf16_kernel(const float* __restrict__ src,
                                unsigned short* __restrict__ dst, long n) {
  long i = (long)blockIdx.x * blockDim.x + threadIdx.x;
  if (i < n) dst[i] = bf16u(src[i]);
}

// ---------------- graph statistics (one block = one graph) ----------------
__global__ void graph_stats_kernel(const float* __restrict__ x,
                                   float* __restrict__ xglob) {
  __shared__ float red[128];
  const int g = blockIdx.x, t = threadIdx.x;
  float v[F_IN];
#pragma unroll
  for (int f = 0; f < F_IN; ++f) v[f] = x[(size_t)(g * NPG + t) * F_IN + f];

  for (int f = 0; f < F_IN; ++f) {
    red[t] = v[f]; __syncthreads();
    for (int s = 64; s > 0; s >>= 1) { if (t < s) red[t] += red[t + s]; __syncthreads(); }
    float mean = red[0] / (float)NPG; __syncthreads();
    red[t] = v[f] * v[f]; __syncthreads();
    for (int s = 64; s > 0; s >>= 1) { if (t < s) red[t] += red[t + s]; __syncthreads(); }
    float var = fabsf(red[0] / (float)NPG - mean * mean); __syncthreads();
    red[t] = v[f]; __syncthreads();
    for (int s = 64; s > 0; s >>= 1) { if (t < s) red[t] = fmaxf(red[t], red[t + s]); __syncthreads(); }
    float mx = red[0]; __syncthreads();
    red[t] = v[f]; __syncthreads();
    for (int s = 64; s > 0; s >>= 1) { if (t < s) red[t] = fminf(red[t], red[t + s]); __syncthreads(); }
    float mn = red[0]; __syncthreads();
    if (t == 0) {
      xglob[g * 24 + f]      = mean;
      xglob[g * 24 + 6 + f]  = var;
      xglob[g * 24 + 12 + f] = mx;
      xglob[g * 24 + 18 + f] = mn;
    }
    __syncthreads();
  }
}

// ---------------- edge features (round 1): builds A1 (E x 19) + mask1 ------
__global__ void edge_feat1_kernel(const float* __restrict__ x,
                                  const int* __restrict__ ei,
                                  const float* __restrict__ bg, const float* __restrict__ bb,
                                  const float* __restrict__ bm, const float* __restrict__ bv,
                                  float* __restrict__ A1, float* __restrict__ mask1) {
  int e = blockIdx.x * blockDim.x + threadIdx.x;
  if (e >= N_EDGES) return;
  int s = ei[2 * e], r = ei[2 * e + 1];
  float xs[F_IN], xr[F_IN], df[F_IN];
#pragma unroll
  for (int f = 0; f < F_IN; ++f) {
    xs[f] = x[(size_t)s * F_IN + f];
    xr[f] = x[(size_t)r * F_IN + f];
    df[f] = xr[f] - xs[f];
  }
  mask1[e] = (xs[3] <= xr[3]) ? 1.f : 0.f;
  float d = sqrtf(df[0] * df[0] + df[1] * df[1] + df[2] * df[2]);
  float inv = (d > 0.f) ? 1.f / d : 0.f;
  float eraw[7] = { df[3], df[4], df[5], d, df[0] * inv, df[1] * inv, df[2] * inv };
#pragma unroll
  for (int j = 0; j < 7; ++j)
    A1[(size_t)e * 19 + 12 + j] =
        (eraw[j] - bm[j]) * bg[j] * rsqrtf(bv[j] + BN_EPS) + bb[j];
#pragma unroll
  for (int f = 0; f < F_IN; ++f) {
    A1[(size_t)e * 19 + f]     = xr[f];
    A1[(size_t)e * 19 + 6 + f] = xs[f];
  }
}

// ---------------- edge features (round 2): e2 (E x 65) + mask2 ------------
__global__ void edge_feat2_kernel(const float* __restrict__ h,
                                  const int* __restrict__ ei,
                                  const float* __restrict__ mask1,
                                  float* __restrict__ e2, float* __restrict__ mask2) {
  int e = blockIdx.x * blockDim.x + threadIdx.x;
  if (e >= N_EDGES) return;
  int s = ei[2 * e], r = ei[2 * e + 1];
  float d0 = h[(size_t)r * H + 0] - h[(size_t)s * H + 0];
  float d1 = h[(size_t)r * H + 1] - h[(size_t)s * H + 1];
  float d2 = h[(size_t)r * H + 2] - h[(size_t)s * H + 2];
  float d = sqrtf(d0 * d0 + d1 * d1 + d2 * d2);
  float inv = (d > 0.f) ? 1.f / d : 0.f;
  mask2[e] = (mask1[e] > 0.f && h[(size_t)s * H + 3] <= h[(size_t)r * H + 3]) ? 1.f : 0.f;
  for (int f = 3; f < H; ++f)
    e2[(size_t)e * 65 + (f - 3)] = h[(size_t)r * H + f] - h[(size_t)s * H + f];
  e2[(size_t)e * 65 + 61] = d;
  e2[(size_t)e * 65 + 62] = d0 * inv;
  e2[(size_t)e * 65 + 63] = d1 * inv;
  e2[(size_t)e * 65 + 64] = d2 * inv;
}

// ---------------- elementwise / scatter helpers ---------------------------
__global__ void zero_kernel(float* p, long n) {
  long i = (long)blockIdx.x * blockDim.x + threadIdx.x;
  if (i < n) p[i] = 0.f;
}
__global__ void relu_kernel(float* p, long n) {
  long i = (long)blockIdx.x * blockDim.x + threadIdx.x;
  if (i < n) p[i] = fmaxf(p[i], 0.f);
}
__global__ void bn_kernel(float* z, const float* g, const float* b,
                          const float* m, const float* v, long n, int D) {
  long i = (long)blockIdx.x * blockDim.x + threadIdx.x;
  if (i >= n) return;
  int c = (int)(i % D);
  z[i] = (z[i] - m[c]) * g[c] * rsqrtf(v[c] + BN_EPS) + b[c];
}
__global__ void scatter_edge_kernel(const float* __restrict__ src,
                                    const int* __restrict__ ei,
                                    const float* __restrict__ fm,
                                    float* __restrict__ dst, int D) {
  long i = (long)blockIdx.x * blockDim.x + threadIdx.x;
  if (i >= (long)N_EDGES * D) return;
  int e = (int)(i / D), c = (int)(i % D);
  float f = fm[e];
  if (f != 0.f) atomicAdd(&dst[(size_t)ei[2 * e + 1] * D + c], src[(size_t)e * D + c] * f);
}
__global__ void scatter_send_kernel(const float* __restrict__ h,
                                    const int* __restrict__ ei,
                                    const float* __restrict__ fm,
                                    float* __restrict__ dst, int D) {
  long i = (long)blockIdx.x * blockDim.x + threadIdx.x;
  if (i >= (long)N_EDGES * D) return;
  int e = (int)(i / D), c = (int)(i % D);
  float f = fm[e];
  if (f != 0.f) atomicAdd(&dst[(size_t)ei[2 * e + 1] * D + c], h[(size_t)ei[2 * e] * D + c] * f);
}
__global__ void cnt_kernel(const int* __restrict__ ei, const float* __restrict__ fm,
                           float* __restrict__ cnt) {
  int e = blockIdx.x * blockDim.x + threadIdx.x;
  if (e >= N_EDGES) return;
  if (fm[e] != 0.f) atomicAdd(&cnt[ei[2 * e + 1]], fm[e]);
}
__global__ void rownorm_relu_kernel(const float* __restrict__ z,
                                    float* __restrict__ hout, int rows, int D) {
  int n = blockIdx.x * blockDim.x + threadIdx.x;
  if (n >= rows) return;
  float ss = 0.f;
  for (int c = 0; c < D; ++c) { float v = z[(size_t)n * D + c]; ss += v * v; }
  float r = rsqrtf(fmaxf(ss, 1e-12f));
  for (int c = 0; c < D; ++c)
    hout[(size_t)n * D + c] = fmaxf(z[(size_t)n * D + c] * r, 0.f);
}
__global__ void neigh_concat_kernel(const float* __restrict__ h,
                                    const float* __restrict__ nsum,
                                    const float* __restrict__ cnt,
                                    float* __restrict__ zin, int rows, int D) {
  long i = (long)blockIdx.x * blockDim.x + threadIdx.x;
  if (i >= (long)rows * 2 * D) return;
  int n = (int)(i / (2 * D)), c = (int)(i % (2 * D));
  float v;
  if (c < D) v = h[(size_t)n * D + c];
  else {
    float cn = cnt[n];
    v = (cn > 0.f) ? nsum[(size_t)n * D + (c - D)] / fmaxf(cn, 1.f) : 0.f;
  }
  zin[i] = v;
}
__global__ void pool_kernel(const float* __restrict__ hfin,
                            const float* __restrict__ xglob,
                            float* __restrict__ z0) {
  int g = blockIdx.x, c = threadIdx.x;  // 256 threads, 256 features
  float mx = -3.4e38f, sm = 0.f;
  for (int i = 0; i < NPG; ++i) {
    float v = hfin[(size_t)(g * NPG + i) * 256 + c];
    mx = fmaxf(mx, v);
    sm += v;
  }
  z0[(size_t)g * 792 + c]       = mx;
  z0[(size_t)g * 792 + 256 + c] = sm / (float)NPG;
  z0[(size_t)g * 792 + 512 + c] = sm;
  if (c < 24) z0[(size_t)g * 792 + 768 + c] = xglob[g * 24 + c];
}
__global__ void heads_final_kernel(const float* __restrict__ hl,
                                   const float* __restrict__ ha,
                                   const float* __restrict__ hsg,
                                   const float* __restrict__ Wlo, const float* __restrict__ blo,
                                   const float* __restrict__ Wao, const float* __restrict__ bao,
                                   const float* __restrict__ Wso, const float* __restrict__ bso,
                                   const float* __restrict__ Wsc, const float* __restrict__ bsc,
                                   float* __restrict__ out) {
  int g = blockIdx.x * blockDim.x + threadIdx.x;
  if (g >= N_GRAPHS) return;
  float loge = blo[0], a0 = bao[0], a1 = bao[1], s0 = bso[0], s1 = bso[1];
  for (int f = 0; f < H; ++f) {
    float vl = hl[(size_t)g * H + f], va = ha[(size_t)g * H + f], vs = hsg[(size_t)g * H + f];
    loge += vl * Wlo[f];
    a0 += va * Wao[f * 2 + 0];  a1 += va * Wao[f * 2 + 1];
    s0 += vs * Wso[f * 2 + 0];  s1 += vs * Wso[f * 2 + 1];
  }
  float t0 = a0 * Wsc[0] + a1 * Wsc[2] + bsc[0];
  float t1 = a0 * Wsc[1] + a1 * Wsc[3] + bsc[1];
  float zen = 1.f / (1.f + expf(-t0));
  float azi = 1.f / (1.f + expf(-t1));
  out[(size_t)g * 5 + 0] = loge;
  out[(size_t)g * 5 + 1] = zen * PI_F;
  out[(size_t)g * 5 + 2] = azi * 2.f * PI_F;
  out[(size_t)g * 5 + 3] = fabsf(s0) + EPSV;
  out[(size_t)g * 5 + 4] = fabsf(s1) + EPSV;
}

// =====================================================================
// host orchestration
// =====================================================================
static inline void gemm(const float* A, const float* W, const float* b, float* C,
                        int M, int N, int K, int act, hipStream_t s) {
  dim3 grid((N + 63) / 64, (M + 15) / 16);
  gemm_bias_act<<<grid, 128, 0, s>>>(A, W, b, C, M, N, K, act);
}
static inline void zero(float* p, long n, hipStream_t s) {
  zero_kernel<<<(int)((n + 255) / 256), 256, 0, s>>>(p, n);
}

extern "C" void kernel_launch(void* const* d_in, const int* in_sizes, int n_in,
                              void* d_out, int out_size, void* d_ws, size_t ws_size,
                              hipStream_t stream) {
  (void)in_sizes; (void)n_in; (void)out_size; (void)ws_size;
  const float* x  = (const float*)d_in[0];
  const int*   ei = (const int*)d_in[1];
  auto P = [&](int i) { return (const float*)d_in[i]; };
  const float *bn_e_g = P(3), *bn_e_b = P(4), *bn_e_m = P(5), *bn_e_v = P(6);
  const float *Wm1 = P(7),  *bm1 = P(8),  *Wm2 = P(9),  *bm2 = P(10);
  const float *Wu1 = P(11), *bu1 = P(12), *Wu2 = P(13), *bu2 = P(14);
  const float *Wk1 = P(15), *bk1 = P(16), *Wk2 = P(17), *bk2 = P(18);
  const float *Wk3 = P(19), *bk3 = P(20), *Wk4 = P(21), *bk4 = P(22);
  const float *Wroot = P(23), *becc = P(24);
  const float *Ws1 = P(25), *bs1 = P(26), *Ws2 = P(27), *bs2 = P(28);
  const float *Wd1 = P(29), *bd1 = P(30), *bn1g = P(31), *bn1b = P(32), *bn1m = P(33), *bn1v = P(34);
  const float *Wd2 = P(35), *bd2 = P(36), *bn2g = P(37), *bn2b = P(38), *bn2m = P(39), *bn2v = P(40);
  const float *Wd3 = P(41), *bd3 = P(42), *bn3g = P(43), *bn3b = P(44), *bn3m = P(45), *bn3v = P(46);
  const float *Wl1 = P(47), *bl1 = P(48), *Wl2 = P(49), *bl2 = P(50), *Wlo = P(51), *blo = P(52);
  const float *Wa1 = P(53), *ba1 = P(54), *Wa2 = P(55), *ba2 = P(56), *Wao = P(57), *bao = P(58);
  const float *Wg1 = P(59), *bg1 = P(60), *Wg2 = P(61), *bg2 = P(62), *Wgo = P(63), *bgo = P(64);
  const float *Wsc = P(65), *bsc = P(66);
  float* out = (float*)d_out;

  // ---------------- workspace pools (reused along the timeline) ----------
  char* ws = (char*)d_ws;
  const size_t MBy = 1ull << 20;
  float* xglob = (float*)(ws + 0);                 //  12 KB
  float* mask1 = (float*)(ws + 64 * 1024);         // 192 KB
  float* mask2 = (float*)(ws + 320 * 1024);        // 192 KB
  float* cnt_e = (float*)(ws + 576 * 1024);        //  64 KB
  char*  SM    = ws + 704 * 1024;                  // small region
  float* z0  = (float*)(SM + 0);                   // 128x792
  float* z1  = (float*)(SM + 512 * 1024);          // 128x2048
  float* z2  = (float*)(SM + 1664 * 1024);         // 128x1024
  float* z3  = (float*)(SM + 2304 * 1024);         // 128x512
  float* ht0 = (float*)(SM + 2624 * 1024);         // 6 x 128x64
  float* ht1 = ht0 + 128 * 64;
  float* ht2 = ht1 + 128 * 64;
  float* ht3 = ht2 + 128 * 64;
  float* ht4 = ht3 + 128 * 64;
  float* ht5 = ht4 + 128 * 64;
  unsigned short* Wk4bf = (unsigned short*)(SM + 3 * MBy); // 512 KB bf16 Wk4
  char* base = ws + 8 * MBy;
  float* hbuf  = (float*)(base);                   // 5 MB : h (N x 64)
  float* h2buf = (float*)(base + 5 * MBy);         // 5 MB : root / h2 (N x 64)
  float* nsum  = (float*)(base + 10 * MBy);        // 9 MB : nsum (N x <=128)
  char*  BIG0  = base + 19 * MBy;                  // 17 MB: A1 -> e2 -> zin
  char*  BIG3  = base + 36 * MBy;                  // 13 MB: agg -> k3 -> hs1
  char*  BIG2  = base + 49 * MBy;                  // 26 MB: M2  -> k2 -> hs2
  char*  BIG1  = base + 75 * MBy;                  // 51 MB: M1 -> Hu1 -> k1 -> zbuf
  // total ~134 MB

  const int E = N_EDGES, N = N_NODES, G = N_GRAPHS;

  // 0) pre-convert Wk4 to bf16 (L2-resident weight for the fused kernel)
  cvt_bf16_kernel<<<(int)((64L * 4096 + 255) / 256), 256, 0, stream>>>(Wk4, Wk4bf, 64L * 4096);

  // 1) per-graph stats
  graph_stats_kernel<<<G, 128, 0, stream>>>(x, xglob);

  // 2) edge features + BN -> A1 (E x 19)
  float* A1 = (float*)BIG0;
  edge_feat1_kernel<<<(E + 255) / 256, 256, 0, stream>>>(x, ei, bn_e_g, bn_e_b,
                                                         bn_e_m, bn_e_v, A1, mask1);
  // 3) edge MLP (WMMA)
  float* M1 = (float*)BIG1;                        // E x 256
  gemm(A1, Wm1, bm1, M1, E, 256, 19, 1, stream);
  float* M2 = (float*)BIG2;                        // E x 128
  gemm(M1, Wm2, bm2, M2, E, 128, 256, 1, stream);

  // 4) scatter to nodes
  float* agg = (float*)BIG3;                       // N x 128
  zero(agg, (long)N * 128, stream);
  scatter_edge_kernel<<<(int)(((long)E * 128 + 255) / 256), 256, 0, stream>>>(M2, ei, mask1, agg, 128);

  // 5) node MLP
  float* Hu1 = (float*)BIG1;                       // N x 128
  gemm(agg, Wu1, bu1, Hu1, N, 128, 128, 1, stream);
  gemm(Hu1, Wu2, bu2, hbuf, N, 64, 128, 1, stream);   // h : N x 64

  // 6) second edge features -> e2 (E x 65)
  float* e2 = (float*)BIG0;
  edge_feat2_kernel<<<(E + 255) / 256, 256, 0, stream>>>(hbuf, ei, mask1, e2, mask2);

  // 7) kernel MLP chain (WMMA)
  float* k1 = (float*)BIG1;
  gemm(e2, Wk1, bk1, k1, E, 64, 65, 1, stream);
  float* k2 = (float*)BIG2;
  gemm(k1, Wk2, bk2, k2, E, 64, 64, 1, stream);
  float* k3 = (float*)BIG3;
  gemm(k2, Wk3, bk3, k3, E, 64, 64, 1, stream);

  // 8) root transform, then fused Wk4-GEMM + einsum + scatter (WMMA + TDM)
  gemm(hbuf, Wroot, becc, h2buf, N, 64, 64, 0, stream);
  kmat_msg_kernel<<<E / 16, 128, 0, stream>>>(k3, hbuf, Wk4bf, bk4, ei, mask2, h2buf);
  relu_kernel<<<(int)(((long)N * 64 + 255) / 256), 256, 0, stream>>>(h2buf, (long)N * 64);

  // 9) edge counts for SAGE mean aggregation
  zero(cnt_e, N, stream);
  cnt_kernel<<<(E + 255) / 256, 256, 0, stream>>>(ei, mask2, cnt_e);

  // 10) SAGE layers
  {
    const int D = 64;
    zero(nsum, (long)N * D, stream);
    scatter_send_kernel<<<(int)(((long)E * D + 255) / 256), 256, 0, stream>>>(h2buf, ei, mask2, nsum, D);
    float* zin  = (float*)BIG0;                    // N x 128
    neigh_concat_kernel<<<(int)(((long)N * 2 * D + 255) / 256), 256, 0, stream>>>(h2buf, nsum, cnt_e, zin, N, D);
    float* zb = (float*)BIG1;                      // N x 128
    gemm(zin, Ws1, bs1, zb, N, 128, 128, 0, stream);
    float* hs1 = (float*)BIG3;                     // N x 128
    rownorm_relu_kernel<<<(N + 127) / 128, 128, 0, stream>>>(zb, hs1, N, 128);

    const int D2 = 128;
    zero(nsum, (long)N * D2, stream);
    scatter_send_kernel<<<(int)(((long)E * D2 + 255) / 256), 256, 0, stream>>>(hs1, ei, mask2, nsum, D2);
    float* zin2 = (float*)BIG0;                    // N x 256
    neigh_concat_kernel<<<(int)(((long)N * 2 * D2 + 255) / 256), 256, 0, stream>>>(hs1, nsum, cnt_e, zin2, N, D2);
    float* zb2 = (float*)BIG1;                     // N x 256
    gemm(zin2, Ws2, bs2, zb2, N, 256, 256, 0, stream);
    float* hs2 = (float*)BIG2;                     // N x 256
    rownorm_relu_kernel<<<(N + 127) / 128, 128, 0, stream>>>(zb2, hs2, N, 256);

    // 11) pooling -> z0 (G x 792)
    pool_kernel<<<G, 256, 0, stream>>>(hs2, xglob, z0);
  }

  // 12) dense stack with leaky-relu + BN (WMMA)
  gemm(z0, Wd1, bd1, z1, G, 2048, 792, 2, stream);
  bn_kernel<<<(int)(((long)G * 2048 + 255) / 256), 256, 0, stream>>>(z1, bn1g, bn1b, bn1m, bn1v, (long)G * 2048, 2048);
  gemm(z1, Wd2, bd2, z2, G, 1024, 2048, 2, stream);
  bn_kernel<<<(int)(((long)G * 1024 + 255) / 256), 256, 0, stream>>>(z2, bn2g, bn2b, bn2m, bn2v, (long)G * 1024, 1024);
  gemm(z2, Wd3, bd3, z3, G, 512, 1024, 2, stream);
  bn_kernel<<<(int)(((long)G * 512 + 255) / 256), 256, 0, stream>>>(z3, bn3g, bn3b, bn3m, bn3v, (long)G * 512, 512);

  // 13) heads (linear chains, WMMA) + final assembly
  gemm(z3, Wl1, bl1, ht0, G, 64, 512, 0, stream);
  gemm(ht0, Wl2, bl2, ht1, G, 64, 64, 0, stream);
  gemm(z3, Wa1, ba1, ht2, G, 64, 512, 0, stream);
  gemm(ht2, Wa2, ba2, ht3, G, 64, 64, 0, stream);
  gemm(z3, Wg1, bg1, ht4, G, 64, 512, 0, stream);
  gemm(ht4, Wg2, bg2, ht5, G, 64, 64, 0, stream);
  heads_final_kernel<<<1, 128, 0, stream>>>(ht1, ht3, ht5, Wlo, blo, Wao, bao,
                                            Wgo, bgo, Wsc, bsc, out);
}
